// ResIndRNN_32272384262325
// MI455X (gfx1250) — compile-verified, matrix operand order checked by reading, not run on hardware
//
#include <hip/hip_runtime.h>

typedef __bf16 bf16_t;
typedef __attribute__((ext_vector_type(16))) __bf16 v16bf;
typedef __attribute__((ext_vector_type(8)))  __bf16 v8bf;
typedef __attribute__((ext_vector_type(8)))  float  v8f;

#define B_  64
#define T_  50
#define IN_ 256
#define H_  2000
#define M_  (B_ * T_)   // 3200
#define KP  2048        // padded K (stride of bf16 activation buffer)
#define NP  2048        // padded N (stride of fp32 GEMM-output buffer)

union FragU { v16bf v; v8bf h[2]; };

// Issue one 16-byte async copy global -> LDS (tracked by ASYNCcnt).
#define ASYNC_CP16(ldsoff, gaddr)                                             \
  asm volatile("global_load_async_to_lds_b128 %0, %1, off"                    \
               :: "v"((unsigned)(ldsoff)),                                    \
                  "v"((unsigned long long)(gaddr))                            \
               : "memory")

// Gather a 16(M/N) x 32(K) bf16 WMMA fragment from an LDS tile (row stride 32
// elements). Per CDNA5 ISA layout: lane L<16 -> row L, K = {0..7} U {16..23};
// lanes 16..31 -> same rows, K = {8..15} U {24..31}.
#define LDS_FRAG(dst, tile, row0)                                             \
  { const int _r  = (row0) + (lane & 15);                                     \
    const int _kb = (lane >> 4) << 3;                                         \
    FragU _f;                                                                 \
    _f.h[0] = *(const v8bf*)&(tile)[_r][_kb];                                 \
    _f.h[1] = *(const v8bf*)&(tile)[_r][_kb + 16];                            \
    (dst) = _f.v; }

// ---------------------------------------------------------------------------
// GEMM: Mout[M_ x NP] (fp32, stride NP) = A[M_ x KP](bf16) * W[NP x KP](bf16)^T
// 256 threads = 8 waves; wave tile 32(M) x 64(N); WG tile 128 x 128.
// Double-buffered async global->LDS staging; WMMA fragments read from LDS.
// ---------------------------------------------------------------------------
__global__ __launch_bounds__(256)
void gemm_bf16_kernel(const bf16_t* __restrict__ A, const bf16_t* __restrict__ W,
                      float* __restrict__ Mout, int kTiles) {
  __shared__ __align__(16) bf16_t sA[2][128][32];   // 16 KB
  __shared__ __align__(16) bf16_t sB[2][128][32];   // 16 KB

  const int tid  = threadIdx.x;
  const int wave = tid >> 5;
  const int lane = tid & 31;
  const int wm = wave & 3;                    // 4 waves along M
  const int wn = wave >> 2;                   // 2 waves along N
  const int mblk = blockIdx.y * 128;
  const int nblk = blockIdx.x * 128;

  // Per-thread copy slice: 2 threads per 64-byte tile row, 2 chunks each.
  const int crow = tid >> 1;                  // 0..127
  const int cb   = (tid & 1) * 16;            // byte offset within row: 0 or 16

  auto issue_tiles = [&](int buf, int kt) {
    const unsigned long long ga =
        (unsigned long long)(A + (size_t)(mblk + crow) * KP + (size_t)kt * 32);
    const unsigned long long gb =
        (unsigned long long)(W + (size_t)(nblk + crow) * KP + (size_t)kt * 32);
    const unsigned la = (unsigned)(uintptr_t)&sA[buf][crow][0];
    const unsigned lb = (unsigned)(uintptr_t)&sB[buf][crow][0];
    ASYNC_CP16(la + cb,      ga + cb);
    ASYNC_CP16(la + cb + 32, ga + cb + 32);
    ASYNC_CP16(lb + cb,      gb + cb);
    ASYNC_CP16(lb + cb + 32, gb + cb + 32);   // 4 async ops / thread / k-step
  };

  v8f acc[2][4];
#pragma unroll
  for (int mi = 0; mi < 2; ++mi)
#pragma unroll
    for (int ni = 0; ni < 4; ++ni)
      acc[mi][ni] = (v8f){0.f, 0.f, 0.f, 0.f, 0.f, 0.f, 0.f, 0.f};

  issue_tiles(0, 0);                          // prologue: fill buffer 0

  for (int kt = 0; kt < kTiles; ++kt) {
    const int cur = kt & 1;
    if (kt + 1 < kTiles) {
      issue_tiles(1 - cur, kt + 1);
      // Wait until only the 4 just-issued copies remain -> current buffer done.
      asm volatile("s_wait_asynccnt 0x4" ::: "memory");
    } else {
      asm volatile("s_wait_asynccnt 0x0" ::: "memory");
    }
    __syncthreads();                          // all waves' copies visible

    v16bf a0, a1, b0v, b1v, b2v, b3v;
    LDS_FRAG(a0,  sA[cur], wm * 32);
    LDS_FRAG(a1,  sA[cur], wm * 32 + 16);
    LDS_FRAG(b0v, sB[cur], wn * 64);
    LDS_FRAG(b1v, sB[cur], wn * 64 + 16);
    LDS_FRAG(b2v, sB[cur], wn * 64 + 32);
    LDS_FRAG(b3v, sB[cur], wn * 64 + 48);

    acc[0][0] = __builtin_amdgcn_wmma_f32_16x16x32_bf16(false, a0, false, b0v, (short)0, acc[0][0], false, false);
    acc[0][1] = __builtin_amdgcn_wmma_f32_16x16x32_bf16(false, a0, false, b1v, (short)0, acc[0][1], false, false);
    acc[0][2] = __builtin_amdgcn_wmma_f32_16x16x32_bf16(false, a0, false, b2v, (short)0, acc[0][2], false, false);
    acc[0][3] = __builtin_amdgcn_wmma_f32_16x16x32_bf16(false, a0, false, b3v, (short)0, acc[0][3], false, false);
    acc[1][0] = __builtin_amdgcn_wmma_f32_16x16x32_bf16(false, a1, false, b0v, (short)0, acc[1][0], false, false);
    acc[1][1] = __builtin_amdgcn_wmma_f32_16x16x32_bf16(false, a1, false, b1v, (short)0, acc[1][1], false, false);
    acc[1][2] = __builtin_amdgcn_wmma_f32_16x16x32_bf16(false, a1, false, b2v, (short)0, acc[1][2], false, false);
    acc[1][3] = __builtin_amdgcn_wmma_f32_16x16x32_bf16(false, a1, false, b3v, (short)0, acc[1][3], false, false);

    __syncthreads();                          // retire buffer before overwrite
  }

  // C/D layout: VGPR r -> row (r + 8*(lane>=16)), lane%16 -> column.
  const int ncol_off = lane & 15;
  const int mrow_off = (lane >> 4) * 8;
  const int m0 = mblk + wm * 32;
  const int n0 = nblk + wn * 64;
#pragma unroll
  for (int mi = 0; mi < 2; ++mi)
#pragma unroll
    for (int ni = 0; ni < 4; ++ni)
#pragma unroll
      for (int r = 0; r < 8; ++r) {
        const int mrow = m0 + mi * 16 + r + mrow_off;
        const int ncol = n0 + ni * 16 + ncol_off;
        Mout[(size_t)mrow * NP + ncol] = acc[mi][ni][r];
      }
}

// ---------------------------------------------------------------------------
// Elementwise / reduction kernels
// ---------------------------------------------------------------------------
__global__ void cvt_x_kernel(const float* __restrict__ x, bf16_t* __restrict__ Abf) {
  int idx = blockIdx.x * blockDim.x + threadIdx.x;   // over M_*KP
  if (idx >= M_ * KP) return;
  int m = idx / KP, k = idx - m * KP;
  float v = (k < IN_) ? x[(size_t)m * IN_ + k] : 0.f;
  Abf[idx] = (bf16_t)v;
}

__global__ void cvt_w_kernel(const float* __restrict__ W, bf16_t* __restrict__ Wbf, int K) {
  int idx = blockIdx.x * blockDim.x + threadIdx.x;   // over NP*KP
  if (idx >= NP * KP) return;
  int n = idx / KP, k = idx - n * KP;
  float v = (n < H_ && k < K) ? W[(size_t)n * K + k] : 0.f;
  Wbf[idx] = (bf16_t)v;
}

__global__ void stats_kernel(const float* __restrict__ Mb,
                             float* __restrict__ mean, float* __restrict__ rstd) {
  int idx = blockIdx.x * blockDim.x + threadIdx.x;   // over T_*H_
  if (idx >= T_ * H_) return;
  int t = idx / H_, h = idx - t * H_;
  float s = 0.f, s2 = 0.f;
  for (int b = 0; b < B_; ++b) {
    float v = Mb[((size_t)(b * T_ + t)) * NP + h];
    s += v; s2 += v * v;
  }
  float mu  = s * (1.f / B_);
  float var = s2 * (1.f / B_) - mu * mu;
  mean[(size_t)t * NP + h] = mu;
  rstd[(size_t)t * NP + h] = rsqrtf(var + 1e-4f);
}

// Fused: BN-normalize + IndRNN scan + residual + fp32 out + bf16 recast.
__global__ void scan_kernel(const float* __restrict__ Mb,
                            const float* __restrict__ mean, const float* __restrict__ rstd,
                            const float* __restrict__ u, const float* __restrict__ g,
                            const float* __restrict__ be,
                            const float* __restrict__ res,   // nullable
                            float* __restrict__ out, bf16_t* __restrict__ Abf) {
  int idx = blockIdx.x * blockDim.x + threadIdx.x;   // over B_*H_
  if (idx >= B_ * H_) return;
  int b = idx / H_, h = idx - b * H_;
  const float uu = u[h], gg = g[h], bb = be[h];
  float hs = 0.f;
  for (int t = 0; t < T_; ++t) {
    const size_t row = (size_t)(b * T_ + t);
    float v = (Mb[row * NP + h] - mean[(size_t)t * NP + h]) * rstd[(size_t)t * NP + h] * gg + bb;
    hs = fmaxf(fmaf(uu, hs, v), 0.f);
    float o = hs;
    if (res) o += res[row * H_ + h];
    out[row * H_ + h] = o;
    Abf[row * KP + h] = (bf16_t)o;                   // next layer's bf16 input
  }
}

// ---------------------------------------------------------------------------
extern "C" void kernel_launch(void* const* d_in, const int* in_sizes, int n_in,
                              void* d_out, int out_size, void* d_ws, size_t ws_size,
                              hipStream_t stream) {
  (void)in_sizes; (void)n_in; (void)out_size; (void)ws_size;
  const float* x   = (const float*)d_in[0];
  const float* W0  = (const float*)d_in[1];
  // d_in[2] = b0 and d_in[7] = bs are mathematically cancelled by BN mean-subtraction.
  const float* u0  = (const float*)d_in[3];
  const float* g0  = (const float*)d_in[4];
  const float* be0 = (const float*)d_in[5];
  const float* Ws  = (const float*)d_in[6];
  const float* us  = (const float*)d_in[8];
  const float* gs  = (const float*)d_in[9];
  const float* bes = (const float*)d_in[10];

  char* ws = (char*)d_ws;
  size_t off = 0;
  auto take = [&](size_t bytes) -> char* {
    char* p = ws + off;
    off = (off + bytes + 255) & ~(size_t)255;
    return p;
  };
  bf16_t* Abf  = (bf16_t*)take((size_t)M_ * KP * 2);   // 13.1 MB
  bf16_t* Wbf  = (bf16_t*)take((size_t)NP * KP * 2);   //  8.4 MB
  float*  Mb   = (float*) take((size_t)M_ * NP * 4);   // 26.2 MB
  float*  mean = (float*) take((size_t)T_ * NP * 4);
  float*  rstd = (float*) take((size_t)T_ * NP * 4);
  float*  actA = (float*) take((size_t)M_ * H_ * 4);   // 25.6 MB
  float*  actB = (float*) take((size_t)M_ * H_ * 4);   // 25.6 MB

  const dim3 ggrid(NP / 128, M_ / 128);                // 16 x 25 workgroups

  auto run_layer = [&](const float* Wptr, int K, int kTiles,
                       const float* u, const float* g, const float* be,
                       const float* res, float* outAct) {
    cvt_w_kernel<<<(NP * KP + 255) / 256, 256, 0, stream>>>(Wptr, Wbf, K);
    gemm_bf16_kernel<<<ggrid, 256, 0, stream>>>(Abf, Wbf, Mb, kTiles);
    stats_kernel<<<(T_ * H_ + 255) / 256, 256, 0, stream>>>(Mb, mean, rstd);
    scan_kernel<<<(B_ * H_ + 255) / 256, 256, 0, stream>>>(Mb, mean, rstd, u, g, be,
                                                           res, outAct, Abf);
  };

  // Stem: K = 256 (8 K-tiles)
  cvt_x_kernel<<<(M_ * KP + 255) / 256, 256, 0, stream>>>(x, Abf);
  run_layer(W0, IN_, IN_ / 32, u0, g0, be0, nullptr, actA);

  // 5 residual blocks of 2 layers; K padded 2000 -> 2048 (zeros), 64 K-tiles.
  for (int blk = 0; blk < 5; ++blk) {
    const int i = 2 * blk;
    run_layer(Ws + (size_t)i * H_ * H_, H_, KP / 32,
              us + (size_t)i * H_, gs + (size_t)i * H_, bes + (size_t)i * H_,
              nullptr, actB);
    float* dst = (blk == 4) ? (float*)d_out : actA;
    run_layer(Ws + (size_t)(i + 1) * H_ * H_, H_, KP / 32,
              us + (size_t)(i + 1) * H_, gs + (size_t)(i + 1) * H_, bes + (size_t)(i + 1) * H_,
              /*res=*/actA, dst);
  }
}
